// LPP49_8409545966409
// MI455X (gfx1250) — matrix operand order repaired and checked by reading, not compiled, not run on hardware
//
#include <hip/hip_runtime.h>

// ---------------------------------------------------------------------------
// Masked log-polar pooling as WMMA GEMM:
//   C[4096x256] = X[4096x50176] * Masks^T, then /counts, reflect-pad, bilerp.
// bf16-split (hi+lo) GEMM with f32 accumulation => ~fp32 accuracy, while the
// 822MB stream of x at 23.3 TB/s (~35us) stays the bottleneck, not compute.
// ---------------------------------------------------------------------------

typedef __attribute__((ext_vector_type(16))) __bf16 v16bf;
typedef __attribute__((ext_vector_type(8)))  float  v8f;

#define HW        50176                    // 224*224
#define NROWS     4096                     // 32*128
#define NM        256                      // pooling bins
#define KCH       1568                     // HW / 32  (K-chunks of 32)
#define KSPLIT    2
#define KC_PER    (KCH / KSPLIT)           // 784 (even)
#define MTILE     64
#define BP_BYTES  ((size_t)NM * HW * 2)    // packed bf16 masks: 25,690,112 B
#define P_ELEMS   ((size_t)NROWS * NM)     // 1,048,576 floats per split
#define OUT_TOTAL (32 * 128 * 56 * 56)     // 12,845,056

// float -> bf16 bits, round-to-nearest-even (valid for normal data)
__device__ __forceinline__ unsigned bf16b(float f) {
  unsigned u = __float_as_uint(f);
  return (u + 0x7FFFu + ((u >> 16) & 1u)) >> 16;
}

// Convert 8 fp32 -> packed hi/lo bf16 (4 dwords each, element order = K order)
__device__ __forceinline__ void cvt8(const float4& a, const float4& b,
                                     uint4& hi, uint4& lo) {
  float f[8] = {a.x, a.y, a.z, a.w, b.x, b.y, b.z, b.w};
  unsigned hb[8], lb[8];
#pragma unroll
  for (int i = 0; i < 8; ++i) {
    hb[i] = bf16b(f[i]);
    float hf = __uint_as_float(hb[i] << 16);
    lb[i] = bf16b(f[i] - hf);
  }
  hi = make_uint4(hb[0] | (hb[1] << 16), hb[2] | (hb[3] << 16),
                  hb[4] | (hb[5] << 16), hb[6] | (hb[7] << 16));
  lo = make_uint4(lb[0] | (lb[1] << 16), lb[2] | (lb[3] << 16),
                  lb[4] | (lb[5] << 16), lb[6] | (lb[7] << 16));
}

// ---------------------------------------------------------------------------
// Pack masks (fp32, 0/1-valued) into bf16 B-fragment layout:
//   block(nblk, kc) = 32 lanes x 16 bf16.  B layout (this code's K order, same as A): lane%16 = N
//   column, lane/16 = K half (0..15 vs 16..31), element e = K within half.
// ---------------------------------------------------------------------------
__global__ void pack_masks_kernel(const float* __restrict__ masks,
                                  unsigned short* __restrict__ Bp) {
  int K = blockIdx.x * 256 + threadIdx.x;   // 0..50175 (exact)
  int m = blockIdx.y;                       // 0..255
  float v = masks[(size_t)m * HW + K];
  int nblk = m >> 4, r = m & 15;
  int kc = K >> 5,  k = K & 31;
  int lane = (k >> 4) * 16 + r;             // B: lane-group = K/16
  int e    = k & 15;
  size_t dst = (((size_t)(nblk * KCH + kc) * 32) + lane) * 16 + e;
  Bp[dst] = (unsigned short)bf16b(v);       // 0x0000 or 0x3F80, exact
}

// ---------------------------------------------------------------------------
// GEMM: grid = (64 M-tiles, 2 K-splits); block = 256 (8 waves).
// WG computes M=64 x N=256; wave w owns N = w*32..w*32+31.
// A tile (64x32 fp32) loaded cooperatively, split hi/lo bf16 ONCE, staged in
// LDS in A-fragment layout (lane%16 = M, K = 16*(lane/16)+e, same order as B).
// ---------------------------------------------------------------------------
__global__ __launch_bounds__(256) void gemm_kernel(
    const float* __restrict__ x, const unsigned short* __restrict__ Bp,
    const float* __restrict__ counts, float* __restrict__ P) {
  // [buf][msub][lane][half] : 2 x 4 x 32 x 2 x 16B = 8KB each
  __shared__ uint4 aHi[2][4][32][2];
  __shared__ uint4 aLo[2][4][32][2];

  const int tid = threadIdx.x, lane = tid & 31, wave = tid >> 5;
  const int rowBase = blockIdx.x * MTILE;
  const int split   = blockIdx.y;
  const int kc0 = split * KC_PER, kc1 = kc0 + KC_PER;

  // Cooperative A-load mapping: thread -> (row am, 8 consecutive K at akq).
  const int am  = tid >> 2;                 // 0..63
  const int akq = (tid & 3) * 8;            // 0,8,16,24
  const int amsub = am >> 4, arr = am & 15;
  const int alane = ((akq >> 4) * 16) + arr;         // A frag lane
  const int ahalf = (akq >> 3) & 1;                  // e0 = 0 or 8 -> 16B half
  const float* aRow = x + (size_t)(rowBase + am) * HW + akq;

  // B fragment bases for this wave's two 16-column blocks
  const unsigned short* bB0 = Bp + ((size_t)(wave * 2 + 0) * KCH) * 512;
  const unsigned short* bB1 = Bp + ((size_t)(wave * 2 + 1) * KCH) * 512;

  v8f acc[4][2] = {};                       // [msub][nsub] f32 16x16 tiles

  for (int kc = kc0; kc < kc1; kc += 2) {
    float4 f00 = *(const float4*)(aRow + (size_t)kc * 32);
    float4 f01 = *(const float4*)(aRow + (size_t)kc * 32 + 4);
    float4 f10 = *(const float4*)(aRow + (size_t)(kc + 1) * 32);
    float4 f11 = *(const float4*)(aRow + (size_t)(kc + 1) * 32 + 4);
    __builtin_prefetch(aRow + (size_t)(kc + 2) * 32, 0, 1);  // global_prefetch

    uint4 h0, l0, h1, l1;
    cvt8(f00, f01, h0, l0);
    cvt8(f10, f11, h1, l1);

    __syncthreads();                        // WAR: prior fragment reads done
    aHi[0][amsub][alane][ahalf] = h0;  aLo[0][amsub][alane][ahalf] = l0;
    aHi[1][amsub][alane][ahalf] = h1;  aLo[1][amsub][alane][ahalf] = l1;
    __syncthreads();

#pragma unroll
    for (int buf = 0; buf < 2; ++buf) {
      const int kcc = kc + buf;
      v16bf b0 = *(const v16bf*)(bB0 + (size_t)kcc * 512 + lane * 16);
      v16bf b1 = *(const v16bf*)(bB1 + (size_t)kcc * 512 + lane * 16);
#pragma unroll
      for (int ms = 0; ms < 4; ++ms) {
        v16bf ah = *(const v16bf*)&aHi[buf][ms][lane][0];
        v16bf al = *(const v16bf*)&aLo[buf][ms][lane][0];
        acc[ms][0] = __builtin_amdgcn_wmma_f32_16x16x32_bf16(
            false, ah, false, b0, (short)0, acc[ms][0], false, false);
        acc[ms][0] = __builtin_amdgcn_wmma_f32_16x16x32_bf16(
            false, al, false, b0, (short)0, acc[ms][0], false, false);
        acc[ms][1] = __builtin_amdgcn_wmma_f32_16x16x32_bf16(
            false, ah, false, b1, (short)0, acc[ms][1], false, false);
        acc[ms][1] = __builtin_amdgcn_wmma_f32_16x16x32_bf16(
            false, al, false, b1, (short)0, acc[ms][1], false, false);
      }
    }
  }

  // Epilogue: C layout -> lane%16 = N, VGPR v -> M = (lane/16)*8 + v.
  // Fold 1/counts here (counts==0 already replaced by 1 on host side).
  const int nbase = wave * 32;
  const int ncol  = lane & 15;
  const int mof   = (lane >> 4) * 8;
  const float ic0 = 1.0f / counts[nbase + ncol];
  const float ic1 = 1.0f / counts[nbase + 16 + ncol];
  float* Pd = P + (size_t)split * P_ELEMS;
#pragma unroll
  for (int ms = 0; ms < 4; ++ms) {
#pragma unroll
    for (int v = 0; v < 8; ++v) {
      int row = rowBase + ms * 16 + mof + v;
      Pd[(size_t)row * NM + nbase + ncol]      = acc[ms][0][v] * ic0;
      Pd[(size_t)row * NM + nbase + 16 + ncol] = acc[ms][1][v] * ic1;
    }
  }
}

// ---------------------------------------------------------------------------
// Sum K-split partials, reflect-pad 16x16 -> 18x18, half-pixel bilinear 56x56.
// padded[i] = orig[ i==0 ? 1 : i==17 ? 14 : i-1 ]  (np.pad reflect by 1)
// ---------------------------------------------------------------------------
__global__ void resize_kernel(const float* __restrict__ P0,
                              const float* __restrict__ P1,
                              float* __restrict__ out) {
  int idx = blockIdx.x * 256 + threadIdx.x;
  if (idx >= OUT_TOTAL) return;
  int ox = idx % 56;
  int t  = idx / 56;
  int oy = t % 56;
  int p  = t / 56;                          // (n*128+c)

  const float s = 18.0f / 56.0f;            // align_corners=False scale
  float sy = (oy + 0.5f) * s - 0.5f;
  float sx = (ox + 0.5f) * s - 0.5f;
  float fy = floorf(sy), fx = floorf(sx);
  float wy = sy - fy,    wx = sx - fx;
  int y0 = (int)fy, x0 = (int)fx;
  int y0c = min(max(y0, 0), 17), y1c = min(max(y0 + 1, 0), 17);
  int x0c = min(max(x0, 0), 17), x1c = min(max(x0 + 1, 0), 17);

  const float* b0 = P0 + (size_t)p * NM;
  const float* b1 = P1 + (size_t)p * NM;
  auto fetch = [&](int i, int j) -> float {
    int ri = (i == 0) ? 1 : ((i == 17) ? 14 : (i - 1));
    int rj = (j == 0) ? 1 : ((j == 17) ? 14 : (j - 1));
    int m = ri * 16 + rj;
    return b0[m] + b1[m];
  };
  float v00 = fetch(y0c, x0c), v01 = fetch(y0c, x1c);
  float v10 = fetch(y1c, x0c), v11 = fetch(y1c, x1c);
  out[idx] = (1.f - wy) * ((1.f - wx) * v00 + wx * v01)
           +        wy  * ((1.f - wx) * v10 + wx * v11);
}

// ---------------------------------------------------------------------------
extern "C" void kernel_launch(void* const* d_in, const int* in_sizes, int n_in,
                              void* d_out, int out_size, void* d_ws,
                              size_t ws_size, hipStream_t stream) {
  (void)in_sizes; (void)n_in; (void)out_size; (void)ws_size;
  const float* x      = (const float*)d_in[0];
  const float* masks  = (const float*)d_in[1];
  const float* counts = (const float*)d_in[2];

  unsigned short* Bp = (unsigned short*)d_ws;              // 25.7 MB
  float* P = (float*)((char*)d_ws + BP_BYTES);             // 2 x 4 MB partials

  dim3 gpack(HW / 256, NM);                                // (196, 256)
  pack_masks_kernel<<<gpack, 256, 0, stream>>>(masks, Bp);

  dim3 ggemm(NROWS / MTILE, KSPLIT);                       // (64, 2)
  gemm_kernel<<<ggemm, 256, 0, stream>>>(x, Bp, counts, P);

  resize_kernel<<<OUT_TOTAL / 256, 256, 0, stream>>>(P, P + P_ELEMS,
                                                     (float*)d_out);
}
